// UltimateIntelligencePiMoE_44942537785871
// MI455X (gfx1250) — compile-verified, matrix operand order checked by reading, not run on hardware
//
#include <hip/hip_runtime.h>
#include <hip/hip_bf16.h>

// ======================= MI455X (gfx1250) implementation ====================
// - All dense matmuls: v_wmma_f32_16x16x32_bf16 (wave32, f32 accumulate).
// - All GEMM weights converted ONCE per call to transposed bf16 (W^T[N][K]),
//   so both A and B fragments are pairs of ds_load_b128 in the documented
//   16-bit WMMA VGPR layout (ISA 7.12.2), concatenated via shufflevector.
// - B tiles are fetched by the Tensor Data Mover (tensor_load_to_lds) with
//   the D# pad feature giving an 80B LDS row pitch (aligned + bank-spread),
//   double-buffered and synchronized with s_wait_tensorcnt.
// - LSTM: x@Wi hoisted into one big GEMM; recurrence runs in 8 persistent
//   workgroups; Wh^T bf16 (64MB) stays resident in the 192MB L2.
// Workspace requirement: ~448MB (see alloc() calls).
// ===========================================================================

typedef __attribute__((ext_vector_type(16))) __bf16 v16bf;
typedef __attribute__((ext_vector_type(8)))  float   v8f;
typedef __attribute__((ext_vector_type(8)))  __bf16  bf16x8;
typedef __attribute__((ext_vector_type(4)))  unsigned int u32x4;
typedef __attribute__((ext_vector_type(8)))  int i32x8;
typedef __attribute__((ext_vector_type(4)))  int i32x4;

#if defined(__has_builtin)
# if __has_builtin(__builtin_amdgcn_tensor_load_to_lds)
#  define HAVE_TDM 1
# endif
#endif
#ifndef HAVE_TDM
# define HAVE_TDM 0
#endif

__device__ __forceinline__ float sigmoidf_(float x) { return 1.0f / (1.0f + __expf(-x)); }

__device__ __forceinline__ v8f wmma_bf16(v16bf a, v16bf b, v8f c) {
  return __builtin_amdgcn_wmma_f32_16x16x32_bf16(false, a, false, b, (short)0, c, false, false);
}

__device__ __forceinline__ v16bf concat8(bf16x8 lo, bf16x8 hi) {
  return __builtin_shufflevector(lo, hi, 0, 1, 2, 3, 4, 5, 6, 7, 8, 9, 10, 11, 12, 13, 14, 15);
}

#if HAVE_TDM
// 2D TDM load of a [tile_y=64 rows][tile_x=32 elems] bf16 tile into LDS.
// D# per ISA ch.8: group0 {count, lds_addr, global_addr, type=2}; group1
// {data_size=1(2B), pad_enable, pad_interval=16 DWORDs(code3), pad_amount=
// 4 DWORDs(code3) -> 80B LDS row pitch, dims, tile dims, dim0 stride}.
__device__ __forceinline__ void tdm_load_tile_bf16(unsigned lds_byte_addr,
                                                   const __bf16* g,
                                                   int tensor_x, int tensor_y,
                                                   long row_stride_elems)
{
  unsigned long ga = (unsigned long)(uintptr_t)g;
  u32x4 g0 = { 1u,                                   // count=1, user D#
               lds_byte_addr,                        // bits 63:32
               (unsigned)(ga & 0xffffffffu),         // global_addr lo
               (unsigned)((ga >> 32) & 0x01ffffffu) | 0x80000000u }; // hi | type=2
  unsigned w0 = (1u << 16) | (1u << 20) | (3u << 22) | (3u << 25);
  i32x8 g1 = {
    (int)w0,
    (int)(((unsigned)tensor_x & 0xffffu) << 16),
    (int)((((unsigned)tensor_x >> 16) & 0xffffu) | (((unsigned)tensor_y & 0xffffu) << 16)),
    (int)((((unsigned)tensor_y >> 16) & 0xffffu) | (32u << 16)),   // tile_dim0=32
    (int)64,                                                       // tile_dim1=64
    (int)(unsigned)(row_stride_elems & 0xffffffffu),
    (int)(unsigned)((row_stride_elems >> 32) & 0xffffu),
    0 };
  i32x4 gz = { 0, 0, 0, 0 };
#if defined(__clang_major__) && (__clang_major__ >= 23)
  i32x8 z8 = { 0, 0, 0, 0, 0, 0, 0, 0 };
  __builtin_amdgcn_tensor_load_to_lds(g0, g1, gz, gz, z8, 0);
#else
  __builtin_amdgcn_tensor_load_to_lds(g0, g1, gz, gz, 0);
#endif
}
#endif

// Generic weight transform: W[batch][K][N] f32 -> WT[batch][N][K] bf16.
__global__ __launch_bounds__(256)
void convT_kernel(const float* __restrict__ src, __bf16* __restrict__ dst,
                  int K, int N, long per)
{
  const long base = (long)blockIdx.y * per;
  const long idx  = (long)blockIdx.x * 256 + threadIdx.x;
  if (idx >= per) return;
  const int k = (int)(idx / N);
  const int n = (int)(idx - (long)k * N);
  dst[base + (long)n * K + k] = (__bf16)src[base + idx];
}

// C[M,N] = act(A[M,K]f32 @ B[K,N] + bias[N]); B given as BT[N][K] bf16.
// 4 waves, 64x64 block tile; B tiles via TDM (double buffered) when available.
__global__ __launch_bounds__(128)
void gemm_bias_act_kernel(const float* __restrict__ A, long sA,
                          const __bf16* __restrict__ BT, long sBT,
                          const float* __restrict__ bias, long sBias,
                          float* __restrict__ C, long sC,
                          int N, int K, int act)
{
  __shared__ __align__(16) __bf16 As[64][40];        // 80B row pitch
  __shared__ __align__(16) __bf16 BsT[2][64][40];    // TDM pad -> 80B pitch
  const int bz = blockIdx.z;
  A  += (long)bz * sA;
  BT += (long)bz * sBT;
  C  += (long)bz * sC;
  const int tid = threadIdx.x, wave = tid >> 5, lane = tid & 31;
  const int lm = lane & 15, half = lane >> 4;
  const int m0 = blockIdx.y * 64, n0 = blockIdx.x * 64;

  v8f acc[4];
  #pragma unroll
  for (int t = 0; t < 4; ++t)
    #pragma unroll
    for (int j = 0; j < 8; ++j) acc[t][j] = 0.0f;

#if HAVE_TDM
  if (wave == 0)
    tdm_load_tile_bf16((unsigned)(uintptr_t)&BsT[0][0][0],
                       BT + (long)n0 * K, K, N, (long)K);
#endif

  for (int k0 = 0; k0 < K; k0 += 32) {
    const int cur = (k0 >> 5) & 1;
    #pragma unroll
    for (int i = tid; i < 64 * 32; i += 128) {       // stage A (f32 -> bf16)
      int r = i >> 5, c = i & 31;
      As[r][c] = (__bf16)A[(long)(m0 + r) * K + (k0 + c)];
    }
#if HAVE_TDM
    if (wave == 0) {
      if (k0 + 32 < K) {
        tdm_load_tile_bf16((unsigned)(uintptr_t)&BsT[cur ^ 1][0][0],
                           BT + (long)n0 * K + (k0 + 32), K, N, (long)K);
        __builtin_amdgcn_s_wait_tensorcnt(1);        // current tile done
      } else {
        __builtin_amdgcn_s_wait_tensorcnt(0);        // drain
      }
    }
#else
    #pragma unroll
    for (int i = tid; i < 64 * 32; i += 128) {       // fallback staging
      int n = i >> 5, c = i & 31;
      BsT[cur][n][c] = BT[(long)(n0 + n) * K + (k0 + c)];
    }
#endif
    if (k0 + 32 < K)                                 // global_prefetch_b8
      __builtin_prefetch(&A[(long)(m0 + (tid >> 1)) * K + k0 + 32], 0, 1);
    __syncthreads();

    v16bf af;
    {
      const bf16x8* ap = (const bf16x8*)&As[wave * 16 + lm][half * 8];
      af = concat8(ap[0], ap[2]);
    }
    #pragma unroll
    for (int nt = 0; nt < 4; ++nt) {
      const bf16x8* bp = (const bf16x8*)&BsT[cur][nt * 16 + lm][half * 8];
      v16bf bfv = concat8(bp[0], bp[2]);
      acc[nt] = wmma_bf16(af, bfv, acc[nt]);
    }
    __syncthreads();
  }

  #pragma unroll
  for (int nt = 0; nt < 4; ++nt) {                   // C: row=v+8*half, col=lm
    const int col = n0 + nt * 16 + lm;
    const float bv = bias ? bias[(long)bz * sBias + col] : 0.0f;
    #pragma unroll
    for (int v = 0; v < 8; ++v) {
      const int row = m0 + wave * 16 + v + 8 * half;
      float x = acc[nt][v] + bv;
      if (act == 1) x = fmaxf(x, 0.0f);
      else if (act == 2) x = sigmoidf_(x);
      else if (act == 3) x = tanhf(x);
      C[(long)row * N + col] = x;
    }
  }
}

// Row-wise LayerNorm in-place; one block per row; gamma/beta per expert.
__global__ __launch_bounds__(256)
void layernorm_kernel(float* __restrict__ xio, const float* __restrict__ g,
                      const float* __restrict__ beta, int len, int rows_per_expert)
{
  const long row = blockIdx.x;
  const int  e   = (int)(row / rows_per_expert);
  float* xr = xio + row * (long)len;
  const float* gr = g    + (long)e * len;
  const float* br = beta + (long)e * len;
  const int tid = threadIdx.x, wave = tid >> 5, lane = tid & 31;
  float s = 0.0f, ss = 0.0f;
  for (int i = tid; i < len; i += 256) { float v = xr[i]; s += v; ss += v * v; }
  #pragma unroll
  for (int m = 16; m >= 1; m >>= 1) { s += __shfl_xor(s, m); ss += __shfl_xor(ss, m); }
  __shared__ float rs[8], rss[8], stats[2];
  if (lane == 0) { rs[wave] = s; rss[wave] = ss; }
  __syncthreads();
  if (tid == 0) {
    float S = 0.0f, SS = 0.0f;
    for (int i = 0; i < 8; ++i) { S += rs[i]; SS += rss[i]; }
    float mean = S / len;
    float var  = SS / len - mean * mean;
    stats[0] = mean; stats[1] = rsqrtf(var + 1e-5f);
  }
  __syncthreads();
  const float mean = stats[0], inv = stats[1];
  for (int i = tid; i < len; i += 256)
    xr[i] = (xr[i] - mean) * inv * gr[i] + br[i];
}

// Self-attention, one block per (head, batch, expert). S=128, DH=64, 4 waves.
__global__ __launch_bounds__(128)
void attention_kernel(const float* __restrict__ qkv, float* __restrict__ o)
{
  const int h = blockIdx.x, bb = blockIdx.y, e = blockIdx.z;
  __shared__ __align__(16) __bf16 Qs[128][64];
  __shared__ __align__(16) __bf16 Ks[128][64];
  __shared__ __align__(16) __bf16 VsT[64][136];      // V transposed, 272B pitch
  __shared__ __align__(16) __bf16 Ps[128][128];
  const int tid = threadIdx.x, wave = tid >> 5, lane = tid & 31;
  const int lm = lane & 15, half = lane >> 4;
  const long base = (long)(e * 8 + bb) * 128 * 3072;

  for (int i = tid; i < 128 * 64; i += 128) {
    int r = i >> 6, c = i & 63;
    long a = base + (long)r * 3072 + h * 64 + c;
    Qs[r][c]  = (__bf16)qkv[a];
    Ks[r][c]  = (__bf16)qkv[a + 1024];
    VsT[c][r] = (__bf16)qkv[a + 2048];
  }
  __syncthreads();

  // scores = Q @ K^T : each wave owns 32 query rows (2 M-tiles x 8 N-tiles)
  v8f sacc[2][8];
  #pragma unroll
  for (int a = 0; a < 2; ++a)
    #pragma unroll
    for (int n = 0; n < 8; ++n)
      #pragma unroll
      for (int j = 0; j < 8; ++j) sacc[a][n][j] = 0.0f;

  #pragma unroll
  for (int kt = 0; kt < 2; ++kt) {
    v16bf af[2];
    #pragma unroll
    for (int mt = 0; mt < 2; ++mt) {
      const bf16x8* qp = (const bf16x8*)&Qs[wave * 32 + mt * 16 + lm][kt * 32 + half * 8];
      af[mt] = concat8(qp[0], qp[2]);
    }
    #pragma unroll
    for (int nt = 0; nt < 8; ++nt) {
      const bf16x8* kp = (const bf16x8*)&Ks[nt * 16 + lm][kt * 32 + half * 8];
      v16bf bfv = concat8(kp[0], kp[2]);
      sacc[0][nt] = wmma_bf16(af[0], bfv, sacc[0][nt]);
      sacc[1][nt] = wmma_bf16(af[1], bfv, sacc[1][nt]);
    }
  }

  // softmax across the 16 lanes of each C-fragment half-group
  const float scale = 0.125f;  // 1/sqrt(64)
  #pragma unroll
  for (int mt = 0; mt < 2; ++mt) {
    #pragma unroll
    for (int v = 0; v < 8; ++v) {
      float mx = -3.0e38f;
      #pragma unroll
      for (int nt = 0; nt < 8; ++nt) mx = fmaxf(mx, sacc[mt][nt][v]);
      #pragma unroll
      for (int m = 8; m >= 1; m >>= 1) mx = fmaxf(mx, __shfl_xor(mx, m));
      float pv[8]; float sum = 0.0f;
      #pragma unroll
      for (int nt = 0; nt < 8; ++nt) {
        pv[nt] = __expf((sacc[mt][nt][v] - mx) * scale);
        sum += pv[nt];
      }
      #pragma unroll
      for (int m = 8; m >= 1; m >>= 1) sum += __shfl_xor(sum, m);
      const float inv = 1.0f / sum;
      const int row = wave * 32 + mt * 16 + v + 8 * half;
      #pragma unroll
      for (int nt = 0; nt < 8; ++nt)
        Ps[row][nt * 16 + lm] = (__bf16)(pv[nt] * inv);
    }
  }
  __syncthreads();

  // O = P @ V  (V fragments now contiguous via VsT)
  v8f oacc[2][4];
  #pragma unroll
  for (int a = 0; a < 2; ++a)
    #pragma unroll
    for (int n = 0; n < 4; ++n)
      #pragma unroll
      for (int j = 0; j < 8; ++j) oacc[a][n][j] = 0.0f;

  #pragma unroll
  for (int kt = 0; kt < 4; ++kt) {
    v16bf af[2];
    #pragma unroll
    for (int mt = 0; mt < 2; ++mt) {
      const bf16x8* pp = (const bf16x8*)&Ps[wave * 32 + mt * 16 + lm][kt * 32 + half * 8];
      af[mt] = concat8(pp[0], pp[2]);
    }
    #pragma unroll
    for (int nt = 0; nt < 4; ++nt) {
      const bf16x8* vp = (const bf16x8*)&VsT[nt * 16 + lm][kt * 32 + half * 8];
      v16bf bfv = concat8(vp[0], vp[2]);
      oacc[0][nt] = wmma_bf16(af[0], bfv, oacc[0][nt]);
      oacc[1][nt] = wmma_bf16(af[1], bfv, oacc[1][nt]);
    }
  }
  #pragma unroll
  for (int mt = 0; mt < 2; ++mt)
    #pragma unroll
    for (int nt = 0; nt < 4; ++nt)
      #pragma unroll
      for (int v = 0; v < 8; ++v) {
        const int row = wave * 32 + mt * 16 + v + 8 * half;
        o[((long)(e * 8 + bb) * 128 + row) * 1024 + h * 64 + nt * 16 + lm] = oacc[mt][nt][v];
      }
}

// Bidirectional LSTM recurrence. One persistent block per (expert, dir).
__global__ __launch_bounds__(256)
void lstm_kernel(const __bf16* __restrict__ whfT,  // [4][4096][1024] bf16
                 const __bf16* __restrict__ whbT,  // [4][4096][1024] bf16
                 const float*  __restrict__ zpre,  // [8][8][128][4096]
                 float* __restrict__ mem)          // [4][8][128][2048]
{
  const int ed = blockIdx.x;            // e*2 + dir
  const int e = ed >> 1, dir = ed & 1;
  __shared__ __align__(16) __bf16 Hs[8][1024];   // 16KB h (bf16)
  __shared__ float Zs[8][4096];                  // 128KB h@Wh this step
  const int tid = threadIdx.x, wave = tid >> 5, lane = tid & 31;
  const int lm = lane & 15, half = lane >> 4;
  const int r  = wave;
  const int lc = lane;

  for (int i = tid; i < 8 * 1024; i += 256) (&Hs[0][0])[i] = (__bf16)0.0f;
  float creg[32];
  #pragma unroll
  for (int j = 0; j < 32; ++j) creg[j] = 0.0f;
  __syncthreads();

  const __bf16* Wbase = (dir ? whbT : whfT) + (long)e * 4096 * 1024;

  for (int t = 0; t < 128; ++t) {
    const int s = dir ? (127 - t) : t;
    #pragma unroll 1
    for (int nt0 = 0; nt0 < 32; ++nt0) {
      const int col0 = wave * 512 + nt0 * 16;
      v8f acc;
      #pragma unroll
      for (int j = 0; j < 8; ++j) acc[j] = 0.0f;
      const __bf16* wrow = Wbase + (long)(col0 + lm) * 1024;
      #pragma unroll 4
      for (int kt = 0; kt < 32; ++kt) {
        v16bf af;
        if (lm < 8) {
          const bf16x8* hp = (const bf16x8*)&Hs[lm][kt * 32 + half * 8];
          af = concat8(hp[0], hp[2]);
        } else {
          #pragma unroll
          for (int j = 0; j < 16; ++j) af[j] = (__bf16)0.0f;   // pad rows
        }
        const bf16x8* wp = (const bf16x8*)(wrow + kt * 32 + half * 8);
        v16bf bfv = concat8(wp[0], wp[2]);
        acc = wmma_bf16(af, bfv, acc);
      }
      if (half == 0) {
        #pragma unroll
        for (int v = 0; v < 8; ++v) Zs[v][col0 + lm] = acc[v];
      }
    }
    __syncthreads();
    const float* zp = zpre + ((long)ed * 8 * 128 + (long)r * 128 + s) * 4096;
    float* mrow = mem + ((long)(e * 8 + r) * 128 + s) * 2048 + dir * 1024;
    #pragma unroll
    for (int j = 0; j < 32; ++j) {
      const int n = lc + 32 * j;
      float zi = Zs[r][n]        + zp[n];
      float zf = Zs[r][n + 1024] + zp[n + 1024];
      float zg = Zs[r][n + 2048] + zp[n + 2048];
      float zo = Zs[r][n + 3072] + zp[n + 3072];
      float c  = sigmoidf_(zf) * creg[j] + sigmoidf_(zi) * tanhf(zg);
      float hv = sigmoidf_(zo) * tanhf(c);
      creg[j]  = c;
      mrow[n]  = hv;
      Hs[r][n] = (__bf16)hv;
    }
    __syncthreads();
  }
}

__global__ __launch_bounds__(256)
void mul_kernel(const float* __restrict__ a, const float* __restrict__ b,
                float* __restrict__ c, long n)
{
  long i = (long)blockIdx.x * 256 + threadIdx.x;
  if (i < n) c[i] = a[i] * b[i];
}

// Token-level router softmax (E=4) + expert combine. One block per token.
__global__ __launch_bounds__(128)
void router_combine_kernel(const float* __restrict__ x, const float* __restrict__ rw,
                           const float* __restrict__ rb, const float* __restrict__ outs,
                           float* __restrict__ out)
{
  const int t = blockIdx.x;
  const int tid = threadIdx.x, wave = tid >> 5, lane = tid & 31;
  const float* xt = x + (long)t * 512;
  float a0 = 0, a1 = 0, a2 = 0, a3 = 0;
  for (int i = tid; i < 512; i += 128) {
    float xv = xt[i];
    a0 += xv * rw[i * 4 + 0]; a1 += xv * rw[i * 4 + 1];
    a2 += xv * rw[i * 4 + 2]; a3 += xv * rw[i * 4 + 3];
  }
  #pragma unroll
  for (int m = 16; m >= 1; m >>= 1) {
    a0 += __shfl_xor(a0, m); a1 += __shfl_xor(a1, m);
    a2 += __shfl_xor(a2, m); a3 += __shfl_xor(a3, m);
  }
  __shared__ float red[4][4];
  __shared__ float wgt[4];
  if (lane == 0) { red[wave][0] = a0; red[wave][1] = a1; red[wave][2] = a2; red[wave][3] = a3; }
  __syncthreads();
  if (tid == 0) {
    float l[4];
    for (int e2 = 0; e2 < 4; ++e2)
      l[e2] = red[0][e2] + red[1][e2] + red[2][e2] + red[3][e2] + rb[e2];
    float mx = fmaxf(fmaxf(l[0], l[1]), fmaxf(l[2], l[3]));
    float sum = 0.0f;
    for (int e2 = 0; e2 < 4; ++e2) { l[e2] = __expf(l[e2] - mx); sum += l[e2]; }
    for (int e2 = 0; e2 < 4; ++e2) wgt[e2] = l[e2] / sum;
  }
  __syncthreads();
  const float w0 = wgt[0], w1 = wgt[1], w2 = wgt[2], w3 = wgt[3];
  const long es = (long)1024 * 512;
  for (int d = tid; d < 512; d += 128) {
    long idx = (long)t * 512 + d;
    out[idx] = w0 * outs[idx] + w1 * outs[idx + es]
             + w2 * outs[idx + 2 * es] + w3 * outs[idx + 3 * es];
  }
}

extern "C" void kernel_launch(void* const* d_in, const int* in_sizes, int n_in,
                              void* d_out, int out_size, void* d_ws, size_t ws_size,
                              hipStream_t stream)
{
  (void)in_sizes; (void)n_in; (void)out_size; (void)ws_size;
  const float* x          = (const float*)d_in[0];
  const float* router_w   = (const float*)d_in[1];
  const float* router_b   = (const float*)d_in[2];
  const float* enc_w1     = (const float*)d_in[3];
  const float* enc_b1     = (const float*)d_in[4];
  const float* enc_w2     = (const float*)d_in[5];
  const float* enc_b2     = (const float*)d_in[6];
  const float* enc_g      = (const float*)d_in[7];
  const float* enc_beta   = (const float*)d_in[8];
  const float* attn_in_w  = (const float*)d_in[9];
  const float* attn_in_b  = (const float*)d_in[10];
  const float* attn_out_w = (const float*)d_in[11];
  const float* attn_out_b = (const float*)d_in[12];
  const float* lstm_wi_f  = (const float*)d_in[13];
  const float* lstm_wh_f  = (const float*)d_in[14];
  const float* lstm_b_f   = (const float*)d_in[15];
  const float* lstm_wi_b  = (const float*)d_in[16];
  const float* lstm_wh_b  = (const float*)d_in[17];
  const float* lstm_b_b   = (const float*)d_in[18];
  const float* sig_w1     = (const float*)d_in[19];
  const float* sig_b1     = (const float*)d_in[20];
  const float* sig_w2     = (const float*)d_in[21];
  const float* sig_b2     = (const float*)d_in[22];
  const float* tan_w1     = (const float*)d_in[23];
  const float* tan_b1     = (const float*)d_in[24];
  const float* tan_w2     = (const float*)d_in[25];
  const float* tan_b2     = (const float*)d_in[26];
  const float* dec_w1     = (const float*)d_in[27];
  const float* dec_b1     = (const float*)d_in[28];
  const float* dec_w2     = (const float*)d_in[29];
  const float* dec_b2     = (const float*)d_in[30];
  const float* dec_g      = (const float*)d_in[31];
  const float* dec_beta   = (const float*)d_in[32];
  float* out = (float*)d_out;

  const int M = 1024;                 // B*S tokens
  size_t off = 0;
  auto alloc = [&](size_t bytes) -> void* {
    void* p = (char*)d_ws + off;
    off += (bytes + 255) & ~(size_t)255;
    return p;
  };
  // transposed bf16 weight arena (232MB)
  auto walloc = [&](long K, long N, int batch) -> __bf16* {
    return (__bf16*)alloc((size_t)batch * K * N * 2);
  };
  __bf16* enc_w1T   = walloc(512, 1024, 4);
  __bf16* enc_w2T   = walloc(1024, 1024, 4);
  __bf16* attn_inT  = walloc(1024, 3072, 4);
  __bf16* attn_outT = walloc(1024, 1024, 4);
  __bf16* wi_fT     = walloc(1024, 4096, 4);
  __bf16* wi_bT     = walloc(1024, 4096, 4);
  __bf16* wh_fT     = walloc(1024, 4096, 4);
  __bf16* wh_bT     = walloc(1024, 4096, 4);
  __bf16* sig_w1T   = walloc(2048, 1024, 4);
  __bf16* sig_w2T   = walloc(1024, 1024, 4);
  __bf16* tan_w1T   = walloc(2048, 1024, 4);
  __bf16* tan_w2T   = walloc(1024, 1024, 4);
  __bf16* dec_w1T   = walloc(1024, 1024, 4);
  __bf16* dec_w2T   = walloc(1024, 512, 4);
  // activations
  float* t1   = (float*)alloc((size_t)4 * M * 1024 * 4);   //  16MB
  float* t2   = (float*)alloc((size_t)4 * M * 1024 * 4);   //  16MB
  float* t3   = (float*)alloc((size_t)4 * M * 1024 * 4);   //  16MB
  float* big  = (float*)alloc((size_t)8 * M * 4096 * 4);   // 128MB (qkv, then zpre)
  float* memb = (float*)alloc((size_t)4 * M * 2048 * 4);   //  32MB
  float* outs = (float*)alloc((size_t)4 * M * 512 * 4);    //   8MB

  auto convT = [&](const float* src, __bf16* dst, int K, int N) {
    long per = (long)K * N;
    convT_kernel<<<dim3((unsigned)((per + 255) / 256), 4), dim3(256), 0, stream>>>(src, dst, K, N, per);
  };
  auto gemm = [&](const float* A, long sA, const __bf16* BT, long sBT,
                  const float* bias, long sBias, float* C, long sC,
                  int N, int K, int act, int batch) {
    dim3 grid(N / 64, M / 64, batch);
    gemm_bias_act_kernel<<<grid, dim3(128), 0, stream>>>(A, sA, BT, sBT, bias, sBias, C, sC, N, K, act);
  };

  // 0) weight transform pass: W[K][N] f32 -> W^T[N][K] bf16
  convT(enc_w1, enc_w1T, 512, 1024);
  convT(enc_w2, enc_w2T, 1024, 1024);
  convT(attn_in_w, attn_inT, 1024, 3072);
  convT(attn_out_w, attn_outT, 1024, 1024);
  convT(lstm_wi_f, wi_fT, 1024, 4096);
  convT(lstm_wi_b, wi_bT, 1024, 4096);
  convT(lstm_wh_f, wh_fT, 1024, 4096);
  convT(lstm_wh_b, wh_bT, 1024, 4096);
  convT(sig_w1, sig_w1T, 2048, 1024);
  convT(sig_w2, sig_w2T, 1024, 1024);
  convT(tan_w1, tan_w1T, 2048, 1024);
  convT(tan_w2, tan_w2T, 1024, 1024);
  convT(dec_w1, dec_w1T, 1024, 1024);
  convT(dec_w2, dec_w2T, 1024, 512);

  // 1) encoder MLP + LN
  gemm(x, 0,               enc_w1T, (long)1024 * 512,  enc_b1, 1024, t1, (long)M * 1024, 1024, 512, 1, 4);
  gemm(t1, (long)M * 1024, enc_w2T, (long)1024 * 1024, enc_b2, 1024, t2, (long)M * 1024, 1024, 1024, 0, 4);
  layernorm_kernel<<<dim3(4 * M), dim3(256), 0, stream>>>(t2, enc_g, enc_beta, 1024, M);

  // 2) self-attention
  gemm(t2, (long)M * 1024, attn_inT, (long)3072 * 1024, attn_in_b, 3072, big, (long)M * 3072, 3072, 1024, 0, 4);
  attention_kernel<<<dim3(16, 8, 4), dim3(128), 0, stream>>>(big, t1);
  gemm(t1, (long)M * 1024, attn_outT, (long)1024 * 1024, attn_out_b, 1024, t3, (long)M * 1024, 1024, 1024, 0, 4);

  // 3) LSTM: hoist x@Wi+b, then the recurrence
  gemm(t3, (long)M * 1024, wi_fT, (long)4096 * 1024, lstm_b_f, 4096, big,                  (long)2 * M * 4096, 4096, 1024, 0, 4);
  gemm(t3, (long)M * 1024, wi_bT, (long)4096 * 1024, lstm_b_b, 4096, big + (long)M * 4096, (long)2 * M * 4096, 4096, 1024, 0, 4);
  lstm_kernel<<<dim3(8), dim3(256), 0, stream>>>(wh_fT, wh_bT, big, memb);

  // 4) gate branches: sigmoid(...) * tanh(...)
  gemm(memb, (long)M * 2048, sig_w1T, (long)1024 * 2048, sig_b1, 1024, t1, (long)M * 1024, 1024, 2048, 1, 4);
  gemm(t1,   (long)M * 1024, sig_w2T, (long)1024 * 1024, sig_b2, 1024, t2, (long)M * 1024, 1024, 1024, 2, 4);
  gemm(memb, (long)M * 2048, tan_w1T, (long)1024 * 2048, tan_b1, 1024, t1, (long)M * 1024, 1024, 2048, 1, 4);
  gemm(t1,   (long)M * 1024, tan_w2T, (long)1024 * 1024, tan_b2, 1024, t3, (long)M * 1024, 1024, 1024, 3, 4);
  mul_kernel<<<dim3(16384), dim3(256), 0, stream>>>(t2, t3, t1, (long)4 * M * 1024);

  // 5) decoder MLP + LN
  gemm(t1, (long)M * 1024, dec_w1T, (long)1024 * 1024, dec_b1, 1024, t2,   (long)M * 1024, 1024, 1024, 1, 4);
  gemm(t2, (long)M * 1024, dec_w2T, (long)512 * 1024,  dec_b2, 512,  outs, (long)M * 512,  512,  1024, 0, 4);
  layernorm_kernel<<<dim3(4 * M), dim3(256), 0, stream>>>(outs, dec_g, dec_beta, 512, M);

  // 6) router softmax + combine
  router_combine_kernel<<<dim3(M), dim3(128), 0, stream>>>(x, router_w, router_b, outs, out);
}
